// ChordAwareTransformer_45208825757801
// MI455X (gfx1250) — matrix operand order repaired and checked by reading, not compiled, not run on hardware
//
#include <hip/hip_runtime.h>
#include <hip/hip_bf16.h>

typedef __bf16 bf16;
typedef __bf16 v16bf __attribute__((ext_vector_type(16)));
typedef __bf16 v8bf  __attribute__((ext_vector_type(8)));
typedef float  v8f   __attribute__((ext_vector_type(8)));

#define DMODEL 1024
#define NHEAD  16
#define DHEAD  64
#define SEQ    2048
#define BATCH  2
#define ROWS   (BATCH * SEQ)   // 4096
#define DAUG   192             // 3 * DHEAD augmented K-dim
#define PAD    40              // padded row (bf16): 80B = 20 banks, 16B aligned
#define KPAD   200             // padded row (bf16) for 192-col tiles: 400B

// ---------------------------------------------------------------------------
// WMMA + fragment helpers (wave32, 16x16x32 bf16 -> f32)
// ---------------------------------------------------------------------------
__device__ __forceinline__ v8f wmma_bf16(v16bf a, v16bf b, v8f c) {
  return __builtin_amdgcn_wmma_f32_16x16x32_bf16(
      false, a, false, b, (short)0, c, false, false);
}

// A-fragment: 16x32 tile, lane = row (lane%16), half hv = lane/16.
__device__ __forceinline__ v16bf load_a_frag(const bf16* row, int k0, int hv) {
  v8bf lo = *(const v8bf*)(row + k0 + hv * 16);
  v8bf hi = *(const v8bf*)(row + k0 + hv * 16 + 8);
  v16bf a;
#pragma unroll
  for (int i = 0; i < 8; i++) { a[i] = lo[i]; a[8 + i] = hi[i]; }
  return a;
}

// B-fragment: 16 consecutive K of one column
__device__ __forceinline__ v16bf load_b_frag(const bf16* p) {
  v8bf lo = *(const v8bf*)p;
  v8bf hi = *(const v8bf*)(p + 8);
  v16bf b;
#pragma unroll
  for (int i = 0; i < 8; i++) { b[i] = lo[i]; b[8 + i] = hi[i]; }
  return b;
}

// CDNA5 async copy: global -> LDS, 16 bytes per lane, tracked by ASYNCcnt.
__device__ __forceinline__ void async_b128(uint32_t ldsoff, const void* g) {
  asm volatile("global_load_async_to_lds_b128 %0, %1, off"
               :: "v"(ldsoff), "v"(g) : "memory");
}
__device__ __forceinline__ void wait_async0() {
  asm volatile("s_wait_asynccnt 0" ::: "memory");
}
__device__ __forceinline__ uint32_t lds_off(const void* p) {
  return (uint32_t)(uintptr_t)p;   // AS3 offset == low 32 bits of generic ptr
}

// ---------------------------------------------------------------------------
// Converters
// ---------------------------------------------------------------------------
__global__ void k_f32_to_bf16(const float* __restrict__ in, bf16* __restrict__ out, int n) {
  int i = blockIdx.x * blockDim.x + threadIdx.x;
  if (i < n) out[i] = (bf16)in[i];
}

__global__ void k_transpose_w(const float* __restrict__ W, bf16* __restrict__ WT) {
  int i = blockIdx.x * blockDim.x + threadIdx.x;   // i = k*1024 + n
  int k = i >> 10, n = i & 1023;
  WT[(size_t)n * DMODEL + k] = (bf16)W[(size_t)k * DMODEL + n];
}

__global__ void k_chordbass(const float* __restrict__ pitch_pc, const float* __restrict__ bass_pc,
                            const float* __restrict__ Wc, const float* __restrict__ bc,
                            const float* __restrict__ Wb, const float* __restrict__ bb,
                            bf16* __restrict__ Ka) {
  int i = blockIdx.x * blockDim.x + threadIdx.x;   // s*1024 + d
  if (i >= SEQ * DMODEL) return;
  int s = i >> 10, d = i & 1023;
  float cf = bc[d], bf = bb[d];
#pragma unroll
  for (int j = 0; j < 12; j++) {
    cf += pitch_pc[s * 12 + j] * Wc[j * DMODEL + d];
    bf += bass_pc[s * 12 + j] * Wb[j * DMODEL + d];
  }
  int h = d >> 6, dh = d & 63;
  bf16 cb = (bf16)cf, bv = (bf16)bf;
#pragma unroll
  for (int b = 0; b < 2; b++) {
    size_t base = ((size_t)((b * NHEAD + h) * SEQ + s)) * DAUG;
    Ka[base + 64 + dh]  = cb;
    Ka[base + 128 + dh] = bv;
  }
}

// ---------------------------------------------------------------------------
// WMMA GEMM with async-LDS double-buffered staging, 2x-unrolled pipeline.
// Block (256 thr, 8 waves) computes 256x64; wave w does rows [32w, 32w+32).
// ---------------------------------------------------------------------------
__global__ void __launch_bounds__(256)
k_gemm(const bf16* __restrict__ X, const bf16* __restrict__ WT,
       const float* __restrict__ bias, void* __restrict__ out,
       int mode, int sec, float scale) {
  __shared__ bf16 Xs[2][256][PAD];
  __shared__ bf16 Ws[2][64][PAD];

  const int tid = threadIdx.x;
  const int lane = tid & 31, wave = tid >> 5;
  const int hv = lane >> 4, ln16 = lane & 15;
  const int mBlk = blockIdx.x * 256;
  const int n0g  = blockIdx.y * 64;

  v8f acc[2][4];
#pragma unroll
  for (int a = 0; a < 2; a++)
#pragma unroll
    for (int t = 0; t < 4; t++) acc[a][t] = v8f{};

  const uint32_t xsBase = lds_off(&Xs[0][0][0]);
  const uint32_t wsBase = lds_off(&Ws[0][0][0]);

  auto stage = [&](int buf, int k0) {
#pragma unroll
    for (int i = 0; i < 4; i++) {                 // X: 1024 x 16B chunks
      int c = tid + i * 256;
      int row = c >> 2, cb = (c & 3) * 16;
      uint32_t l = xsBase + (uint32_t)(buf * 256 * PAD * 2 + row * PAD * 2 + cb);
      const char* g = (const char*)(X + (size_t)(mBlk + row) * DMODEL + k0) + cb;
      async_b128(l, g);
    }
    {                                              // W: 256 x 16B chunks
      int row = tid >> 2, cb = (tid & 3) * 16;
      uint32_t l = wsBase + (uint32_t)(buf * 64 * PAD * 2 + row * PAD * 2 + cb);
      const char* g = (const char*)(WT + (size_t)(n0g + row) * DMODEL + k0) + cb;
      async_b128(l, g);
    }
  };

  auto compute = [&](int buf) {
    v16bf a0 = load_a_frag(&Xs[buf][wave * 32 + ln16][0], 0, hv);
    v16bf a1 = load_a_frag(&Xs[buf][wave * 32 + 16 + ln16][0], 0, hv);
#pragma unroll
    for (int t = 0; t < 4; t++) {
      v16bf b = load_b_frag(&Ws[buf][t * 16 + ln16][hv * 16]);
      acc[0][t] = wmma_bf16(a0, b, acc[0][t]);
      acc[1][t] = wmma_bf16(a1, b, acc[1][t]);
    }
  };

  const int NT = DMODEL / 32;   // 32 (even)
  stage(0, 0);
  wait_async0();
  __syncthreads();

  // branch-free steady state, 2 steps per iteration
  for (int kt = 0; kt < NT - 2; kt += 2) {
    stage(1, (kt + 1) * 32);
    compute(0);
    wait_async0(); __syncthreads();
    stage(0, (kt + 2) * 32);
    compute(1);
    wait_async0(); __syncthreads();
  }
  // epilogue: last two steps
  stage(1, (NT - 1) * 32);
  compute(0);
  wait_async0(); __syncthreads();
  compute(1);

#pragma unroll
  for (int ms = 0; ms < 2; ms++) {
#pragma unroll
    for (int t = 0; t < 4; t++) {
      int d = n0g + t * 16 + ln16;
      float bv = bias[d];
      int h = d >> 6, dh = d & 63;
#pragma unroll
      for (int i = 0; i < 8; i++) {
        int m = mBlk + wave * 32 + ms * 16 + hv * 8 + i;   // row = b*2048 + s
        float y = (acc[ms][t][i] + bv) * scale;
        if (mode == 2) {
          ((float*)out)[(size_t)m * DMODEL + d] = y;
        } else if (mode == 1) {
          int b = m >> 11, s = m & 2047;
          ((bf16*)out)[((size_t)((b * NHEAD + h) * DHEAD + dh)) * SEQ + s] = (bf16)y;
        } else {
          int b = m >> 11, s = m & 2047;
          ((bf16*)out)[((size_t)((b * NHEAD + h) * SEQ + s)) * DAUG + sec + dh] = (bf16)y;
        }
      }
    }
  }
}

// ---------------------------------------------------------------------------
// Flash attention (augmented 192-dim heads), async-LDS K/V staging,
// 2x-unrolled branch-free pipeline.
// ---------------------------------------------------------------------------
__global__ void __launch_bounds__(256)
k_attn(const bf16* __restrict__ Qa, const bf16* __restrict__ Ka,
       const bf16* __restrict__ Vt, bf16* __restrict__ ctx) {
  __shared__ bf16 Ks[2][32][KPAD];     // [key][192-dim] padded
  __shared__ bf16 Vs[2][64][PAD];      // [dh][32 keys]  padded
  __shared__ bf16 plds[8][16][PAD];    // per-wave P tile [qrow][key]

  const int tid = threadIdx.x;
  const int lane = tid & 31, wave = tid >> 5;
  const int hv = lane >> 4, ln16 = lane & 15;
  const int bh = blockIdx.y;                       // b*16 + h
  const int q0 = (blockIdx.x * 8 + wave) * 16;

  const bf16* qrow = Qa + ((size_t)bh * SEQ + q0 + ln16) * DAUG;
  v16bf qf[6];
#pragma unroll
  for (int c = 0; c < 6; c++) qf[c] = load_a_frag(qrow, c * 32, hv);

  v8f O[4] = {v8f{}, v8f{}, v8f{}, v8f{}};
  float mrow[8], lrow[8];
#pragma unroll
  for (int i = 0; i < 8; i++) { mrow[i] = -3.4e38f; lrow[i] = 0.f; }

  const bf16* Kbase = Ka + (size_t)bh * SEQ * DAUG;
  const bf16* Vbase = Vt + (size_t)bh * DHEAD * SEQ;
  const uint32_t ksBase = lds_off(&Ks[0][0][0]);
  const uint32_t vsBase = lds_off(&Vs[0][0][0]);

  auto stage = [&](int buf, int k0) {
#pragma unroll
    for (int i = 0; i < 3; i++) {                 // K: 768 x 16B chunks
      int c = tid + i * 256;
      int row = c / 24, part = c % 24;
      uint32_t l = ksBase + (uint32_t)(buf * 32 * KPAD * 2 + row * KPAD * 2 + part * 16);
      const char* g = (const char*)(Kbase + (size_t)(k0 + row) * DAUG) + part * 16;
      async_b128(l, g);
    }
    {                                              // V: 256 x 16B chunks
      int row = tid >> 2, cb = (tid & 3) * 16;
      uint32_t l = vsBase + (uint32_t)(buf * 64 * PAD * 2 + row * PAD * 2 + cb);
      const char* g = (const char*)(Vbase + (size_t)row * SEQ + k0) + cb;
      async_b128(l, g);
    }
  };

  auto step = [&](int buf) {
    // ---- scores: two 16x16 tiles (32 keys), K-dim = 192 ----
    v8f S[2] = {v8f{}, v8f{}};
#pragma unroll
    for (int sub = 0; sub < 2; sub++) {
      const bf16* krow = &Ks[buf][sub * 16 + ln16][0];
#pragma unroll
      for (int c = 0; c < 6; c++) {
        v16bf kf = load_b_frag(krow + c * 32 + hv * 16);
        S[sub] = wmma_bf16(qf[c], kf, S[sub]);
      }
    }
    // ---- online softmax (lane = 1 key column, 8 query rows of its half) ----
    float corr[8];
#pragma unroll
    for (int i = 0; i < 8; i++) {
      float mx = fmaxf(S[0][i], S[1][i]);
      mx = fmaxf(mx, __shfl_xor(mx, 1, 32));
      mx = fmaxf(mx, __shfl_xor(mx, 2, 32));
      mx = fmaxf(mx, __shfl_xor(mx, 4, 32));
      mx = fmaxf(mx, __shfl_xor(mx, 8, 32));
      float mnew = fmaxf(mrow[i], mx);
      corr[i] = __expf(mrow[i] - mnew);
      float p0 = __expf(S[0][i] - mnew);
      float p1 = __expf(S[1][i] - mnew);
      S[0][i] = p0; S[1][i] = p1;
      float rs = p0 + p1;
      rs += __shfl_xor(rs, 1, 32);
      rs += __shfl_xor(rs, 2, 32);
      rs += __shfl_xor(rs, 4, 32);
      rs += __shfl_xor(rs, 8, 32);
      lrow[i] = lrow[i] * corr[i] + rs;
      mrow[i] = mnew;
    }
#pragma unroll
    for (int t = 0; t < 4; t++)
#pragma unroll
      for (int i = 0; i < 8; i++) O[t][i] *= corr[i];

    // ---- P -> LDS (C-layout) -> A-fragment reload (intra-wave, DS in-order) ----
#pragma unroll
    for (int sub = 0; sub < 2; sub++)
#pragma unroll
      for (int i = 0; i < 8; i++)
        plds[wave][hv * 8 + i][sub * 16 + ln16] = (bf16)S[sub][i];
    asm volatile("s_wait_dscnt 0" ::: "memory");
    v16bf pf = load_a_frag(&plds[wave][ln16][0], 0, hv);

    // ---- O += P(16x32) x V(32x64) ----
#pragma unroll
    for (int t = 0; t < 4; t++) {
      v16bf vf = load_b_frag(&Vs[buf][t * 16 + ln16][hv * 16]);
      O[t] = wmma_bf16(pf, vf, O[t]);
    }
  };

  const int NT = SEQ / 32;   // 64 (even)
  stage(0, 0);
  wait_async0();
  __syncthreads();

  for (int kt = 0; kt < NT - 2; kt += 2) {
    stage(1, (kt + 1) * 32);
    step(0);
    wait_async0(); __syncthreads();
    stage(0, (kt + 2) * 32);
    step(1);
    wait_async0(); __syncthreads();
  }
  stage(1, (NT - 1) * 32);
  step(0);
  wait_async0(); __syncthreads();
  step(1);

  // ---- finalize + store ctx (bf16 row-major for final GEMM) ----
  int b = bh >> 4, h = bh & 15;
#pragma unroll
  for (int t = 0; t < 4; t++) {
    int dh = t * 16 + ln16;
#pragma unroll
    for (int i = 0; i < 8; i++) {
      int s = q0 + hv * 8 + i;
      float v = O[t][i] / lrow[i];
      ctx[((size_t)(b * SEQ + s)) * DMODEL + h * DHEAD + dh] = (bf16)v;
    }
  }
}

// ---------------------------------------------------------------------------
// Launch
// ---------------------------------------------------------------------------
extern "C" void kernel_launch(void* const* d_in, const int* in_sizes, int n_in,
                              void* d_out, int out_size, void* d_ws, size_t ws_size,
                              hipStream_t stream) {
  const float* x        = (const float*)d_in[0];
  const float* pitch_pc = (const float*)d_in[1];
  const float* bass_pc  = (const float*)d_in[2];
  const float* W[6]  = {(const float*)d_in[3],  (const float*)d_in[5],
                        (const float*)d_in[7],  (const float*)d_in[9],
                        (const float*)d_in[11], (const float*)d_in[13]};
  const float* Bi[6] = {(const float*)d_in[4],  (const float*)d_in[6],
                        (const float*)d_in[8],  (const float*)d_in[10],
                        (const float*)d_in[12], (const float*)d_in[14]};
  const float* Wc = (const float*)d_in[15];
  const float* bc = (const float*)d_in[16];
  const float* Wb = (const float*)d_in[17];
  const float* bb = (const float*)d_in[18];

  char* ws = (char*)d_ws;
  size_t off = 0;
  bf16* xb  = (bf16*)(ws + off); off += (size_t)ROWS * DMODEL * 2;               // 8 MB
  bf16* WT  = (bf16*)(ws + off); off += (size_t)6 * DMODEL * DMODEL * 2;         // 12 MB
  bf16* Qa  = (bf16*)(ws + off); off += (size_t)BATCH * NHEAD * SEQ * DAUG * 2;  // 25 MB
  bf16* Kaw = (bf16*)(ws + off); off += (size_t)BATCH * NHEAD * SEQ * DAUG * 2;  // 25 MB
  bf16* Vt  = (bf16*)(ws + off); off += (size_t)BATCH * NHEAD * DHEAD * SEQ * 2; // 8 MB
  bf16* ctx = (bf16*)(ws + off); off += (size_t)ROWS * DMODEL * 2;               // 8 MB

  {
    int n = ROWS * DMODEL;
    k_f32_to_bf16<<<(n + 255) / 256, 256, 0, stream>>>(x, xb, n);
  }
  for (int z = 0; z < 6; z++) {
    k_transpose_w<<<(DMODEL * DMODEL) / 256, 256, 0, stream>>>(
        W[z], WT + (size_t)z * DMODEL * DMODEL);
  }
  k_chordbass<<<(SEQ * DMODEL) / 256, 256, 0, stream>>>(
      pitch_pc, bass_pc, Wc, bc, Wb, bb, Kaw);

  dim3 gg(ROWS / 256, DMODEL / 64);   // 16 x 16 blocks of 256
  const float qscale = 1.0f / 24.0f;  // (1/sqrt(64)) / 3 folded into Q
  k_gemm<<<gg, 256, 0, stream>>>(xb, WT + 0 * (size_t)DMODEL * DMODEL, Bi[0], Qa, 0, 0,   qscale);
  k_gemm<<<gg, 256, 0, stream>>>(xb, WT + 1 * (size_t)DMODEL * DMODEL, Bi[1], Qa, 0, 64,  qscale);
  k_gemm<<<gg, 256, 0, stream>>>(xb, WT + 2 * (size_t)DMODEL * DMODEL, Bi[2], Qa, 0, 128, qscale);
  k_gemm<<<gg, 256, 0, stream>>>(xb, WT + 3 * (size_t)DMODEL * DMODEL, Bi[3], Kaw, 0, 0,  1.0f);
  k_gemm<<<gg, 256, 0, stream>>>(xb, WT + 4 * (size_t)DMODEL * DMODEL, Bi[4], Vt,  1, 0,  1.0f);

  k_attn<<<dim3(SEQ / 128, BATCH * NHEAD), 256, 0, stream>>>(Qa, Kaw, Vt, ctx);

  k_gemm<<<gg, 256, 0, stream>>>(ctx, WT + 5 * (size_t)DMODEL * DMODEL, Bi[5], d_out, 2, 0, 1.0f);

  (void)in_sizes; (void)n_in; (void)out_size; (void)ws_size;
}
